// MultiHeadAttention_32478542692474
// MI455X (gfx1250) — matrix operand order repaired and checked by reading, not compiled, not run on hardware
//
#include <hip/hip_runtime.h>
#include <hip/hip_bf16.h>

// MHA for B=4, S=2048, D=512, H=8, DK=64 on gfx1250 (wave32, WMMA bf16,
// double-buffered async global->LDS staging for K/V in the attention stage).

typedef __attribute__((ext_vector_type(16))) __bf16 v16bf;
typedef __attribute__((ext_vector_type(8)))  float  v8f;

#define Bn  4
#define Sn  2048
#define Dn  512
#define Hn  8
#define DKn 64

__device__ __forceinline__ v8f wmma_bf16(v16bf a, v16bf b, v8f c) {
  // (neg_a, A, neg_b, B, c_mod, C, reuse_a, reuse_b)
  return __builtin_amdgcn_wmma_f32_16x16x32_bf16(false, a, false, b, (short)0, c,
                                                 false, false);
}

// CDNA5 async copy: 16 bytes per lane, global -> LDS, tracked by ASYNCcnt.
__device__ __forceinline__ void async_load_b128(unsigned lds_off,
                                                const void* gptr) {
  asm volatile("global_load_async_to_lds_b128 %0, %1, off"
               :: "v"(lds_off), "v"((unsigned long long)(uintptr_t)gptr)
               : "memory");
}

// ---------------------------------------------------------------------------
// Kernel 0: one-time fp32 -> bf16 conversion of x (q,k,v), Wq, Wout.
// One float4 per thread.
// ---------------------------------------------------------------------------
__global__ __launch_bounds__(256) void mha_convert(
    const float* __restrict__ q, const float* __restrict__ k,
    const float* __restrict__ v, const float* __restrict__ Wq,
    const float* __restrict__ Wout,
    __bf16* __restrict__ xb, __bf16* __restrict__ Wqb,
    __bf16* __restrict__ Woutb)
{
  const size_t NX = (size_t)Bn * Sn * Dn;       // 4,194,304 per tensor
  const size_t NW = (size_t)Hn * DKn * Dn;      // 262,144
  size_t base = ((size_t)blockIdx.x * blockDim.x + threadIdx.x) * 4;

  const float* src;
  __bf16*      dst;
  size_t       off;
  if (base < NX)               { src = q;    dst = xb;          off = base; }
  else if (base < 2 * NX)      { src = k;    dst = xb + NX;     off = base - NX; }
  else if (base < 3 * NX)      { src = v;    dst = xb + 2 * NX; off = base - 2 * NX; }
  else if (base < 3 * NX + NW) { src = Wq;   dst = Wqb;         off = base - 3 * NX; }
  else                         { src = Wout; dst = Woutb;       off = base - 3 * NX - NW; }

  float4 fv = *reinterpret_cast<const float4*>(src + off);
  dst[off + 0] = (__bf16)fv.x;
  dst[off + 1] = (__bf16)fv.y;
  dst[off + 2] = (__bf16)fv.z;
  dst[off + 3] = (__bf16)fv.w;
}

// ---------------------------------------------------------------------------
// Kernel 1: per-head projection  xh = x @ Wq[h]^T + bq[h]   (all-bf16 GEMM)
// One wave computes a 16(S) x 64(DK) tile for one of {q,k,v}.
// qh gets the softmax scale folded in. v stored transposed vt [b,h,e,s].
// ---------------------------------------------------------------------------
__global__ __launch_bounds__(256) void mha_proj(
    const __bf16* __restrict__ xb, const __bf16* __restrict__ Wqb,
    const float* __restrict__ bq,
    __bf16* __restrict__ qh, __bf16* __restrict__ kh, __bf16* __restrict__ vt)
{
  const size_t NX = (size_t)Bn * Sn * Dn;

  int w    = blockIdx.x * 8 + (threadIdx.x >> 5);
  int lane = threadIdx.x & 31;
  int half = lane >> 4;
  int l16  = lane & 15;

  int which = w % 3;            w /= 3;
  int stile = w % (Sn / 16);    w /= (Sn / 16);
  int h     = w % Hn;
  int b     = w / Hn;
  int s0    = stile * 16;

  const __bf16* arow = xb + which * NX + ((size_t)b * Sn + s0 + l16) * Dn;

  v8f zero = {};
  v8f acc[4];
  #pragma unroll
  for (int nt = 0; nt < 4; ++nt) acc[nt] = zero;

  for (int k0 = 0; k0 < Dn; k0 += 32) {
    // A fragment: row = s0+l16, K = {0..15} / {16..31} per half
    v16bf a;
    #pragma unroll
    for (int j = 0; j < 8; ++j) {
      a[j]     = arow[k0 + half * 16 + j];
      a[8 + j] = arow[k0 + half * 16 + 8 + j];
    }
    #pragma unroll
    for (int nt = 0; nt < 4; ++nt) {
      const v16bf* wp = (const v16bf*)(Wqb + ((size_t)h * DKn + nt * 16 + l16) * Dn
                                           + k0 + half * 16);
      acc[nt] = wmma_bf16(a, *wp, acc[nt]);
    }
  }

  // epilogue: + bias (scale folded into q projection), store bf16.
  float qs = (which == 0) ? 0.125f : 1.0f;   // 1/sqrt(64) folded into qh
  #pragma unroll
  for (int nt = 0; nt < 4; ++nt) {
    float bv = bq[h * DKn + nt * 16 + l16];
    #pragma unroll
    for (int r = 0; r < 8; ++r) {
      float val = (acc[nt][r] + bv) * qs;
      int   m   = 8 * half + r;
      if (which == 2) {
        vt[((size_t)(b * Hn + h) * DKn + nt * 16 + l16) * Sn + s0 + m] =
            (__bf16)val;
      } else {
        __bf16* dst = (which == 0) ? qh : kh;
        dst[((size_t)(b * Hn + h) * Sn + s0 + m) * DKn + nt * 16 + l16] =
            (__bf16)val;
      }
    }
  }
}

// ---------------------------------------------------------------------------
// Kernel 2: flash attention. One block owns (b,h) x 128 queries (8 waves x 16).
// 32-key chunks of K and V^T are staged into double-buffered LDS with async
// global->LDS copies; chunk i+1 is prefetched while chunk i is computed
// (wait ASYNCcnt <= 2 keeps the newest pair in flight). Online softmax,
// P transposed through per-wave LDS, PV with WMMA.
// ---------------------------------------------------------------------------
__global__ __launch_bounds__(256) void mha_attn(
    const __bf16* __restrict__ qh, const __bf16* __restrict__ kh,
    const __bf16* __restrict__ vt, __bf16* __restrict__ ctx)
{
  __shared__ __attribute__((aligned(128))) __bf16 lds_k[2][32][64];   // [buf][key][e]
  __shared__ __attribute__((aligned(128))) __bf16 lds_vt[2][64][32];  // [buf][e][key]
  __shared__ __attribute__((aligned(128))) __bf16 lds_p[8][16][36];

  int tid  = threadIdx.x;
  int wid  = tid >> 5;
  int lane = tid & 31;
  int half = lane >> 4;
  int l16  = lane & 15;

  int w    = blockIdx.x;
  int qblk = w % (Sn / 128);  w /= (Sn / 128);
  int h    = w % Hn;
  int b    = w / Hn;
  int s0   = qblk * 128 + wid * 16;
  size_t bh = (size_t)(b * Hn + h);

  // cooperative staging: 16 bytes (8 bf16) per thread per buffer
  int krow = tid >> 3, kcol = (tid & 7) * 8;   // lds_k  : 32x64
  int vrow = tid >> 2, vcol = (tid & 3) * 8;   // lds_vt : 64x32
  unsigned lk_off[2], lv_off[2];
  #pragma unroll
  for (int buf = 0; buf < 2; ++buf) {
    lk_off[buf] = (unsigned)(uintptr_t)&lds_k[buf][krow][kcol];
    lv_off[buf] = (unsigned)(uintptr_t)&lds_vt[buf][vrow][vcol];
  }
  const __bf16* kg0 = kh + (bh * Sn + krow) * DKn + kcol;   // + kc*DKn
  const __bf16* vg0 = vt + (bh * DKn + vrow) * Sn + vcol;   // + kc

  // Q A-fragments (scale already folded in at projection)
  v16bf qf[2];
  #pragma unroll
  for (int ks = 0; ks < 2; ++ks) {
    const __bf16* qrow = qh + (bh * Sn + s0 + l16) * DKn + ks * 32;
    #pragma unroll
    for (int j = 0; j < 8; ++j) {
      qf[ks][j]     = qrow[half * 16 + j];
      qf[ks][8 + j] = qrow[half * 16 + 8 + j];
    }
  }

  v8f zero = {};
  v8f oacc[4];
  float mrun[8], lrun[8];
  #pragma unroll
  for (int nt = 0; nt < 4; ++nt) oacc[nt] = zero;
  #pragma unroll
  for (int r = 0; r < 8; ++r) { mrun[r] = -3.0e38f; lrun[r] = 0.f; }

  // prologue: stage chunk 0 into buffer 0
  async_load_b128(lk_off[0], kg0);
  async_load_b128(lv_off[0], vg0);

  for (int kc = 0; kc < Sn; kc += 32) {
    int cur = (kc >> 5) & 1;
    if (kc + 32 < Sn) {
      // prefetch next chunk into the other buffer (its readers finished at
      // the bottom barrier of the previous iteration)
      async_load_b128(lk_off[cur ^ 1], kg0 + (size_t)(kc + 32) * DKn);
      async_load_b128(lv_off[cur ^ 1], vg0 + (kc + 32));
      asm volatile("s_wait_asynccnt 0x2" ::: "memory");  // chunk kc landed
    } else {
      asm volatile("s_wait_asynccnt 0x0" ::: "memory");
    }
    __syncthreads();  // chunk kc visible block-wide

    // ---- scores S = Q K^T (16 x 32), two 16x16 accumulators -------------
    v8f sa_acc = zero, sb_acc = zero;
    #pragma unroll
    for (int ks = 0; ks < 2; ++ks) {
      const v16bf* kp0 = (const v16bf*)&lds_k[cur][l16][ks * 32 + half * 16];
      sa_acc = wmma_bf16(qf[ks], *kp0, sa_acc);
      const v16bf* kp1 = (const v16bf*)&lds_k[cur][16 + l16][ks * 32 + half * 16];
      sb_acc = wmma_bf16(qf[ks], *kp1, sb_acc);
    }

    // ---- online softmax (row = 8*half + r, cols across 16-lane half) ----
    #pragma unroll
    for (int r = 0; r < 8; ++r) {
      float sa = sa_acc[r];
      float sb = sb_acc[r];
      float cm = fmaxf(sa, sb);
      #pragma unroll
      for (int off = 1; off < 16; off <<= 1)
        cm = fmaxf(cm, __shfl_xor(cm, off, 32));
      float mnew  = fmaxf(mrun[r], cm);
      float alpha = __expf(mrun[r] - mnew);
      float p0    = __expf(sa - mnew);
      float p1    = __expf(sb - mnew);
      float ps    = p0 + p1;
      #pragma unroll
      for (int off = 1; off < 16; off <<= 1)
        ps += __shfl_xor(ps, off, 32);
      lrun[r] = lrun[r] * alpha + ps;
      mrun[r] = mnew;
      #pragma unroll
      for (int nt = 0; nt < 4; ++nt) oacc[nt][r] *= alpha;
      lds_p[wid][8 * half + r][l16]      = (__bf16)p0;
      lds_p[wid][8 * half + r][16 + l16] = (__bf16)p1;
    }
    asm volatile("s_wait_dscnt 0x0" ::: "memory");  // per-wave P transpose

    // ---- P as A-fragment (16 rows x 32 keys) ----------------------------
    v16bf pa;
    #pragma unroll
    for (int j = 0; j < 8; ++j) {
      pa[j]     = lds_p[wid][l16][half * 16 + j];
      pa[8 + j] = lds_p[wid][l16][half * 16 + 8 + j];
    }

    // ---- O += P @ V  (V^T in LDS: contiguous keys per lane) -------------
    #pragma unroll
    for (int nt = 0; nt < 4; ++nt) {
      const v16bf* vp = (const v16bf*)&lds_vt[cur][nt * 16 + l16][half * 16];
      oacc[nt] = wmma_bf16(pa, *vp, oacc[nt]);
    }

    __syncthreads();  // all waves done with buf[cur]; safe to refill at kc+64
  }

  // ---- normalize and store head output into concat layout ---------------
  #pragma unroll
  for (int nt = 0; nt < 4; ++nt) {
    #pragma unroll
    for (int r = 0; r < 8; ++r) {
      float vv = oacc[nt][r] / lrun[r];
      int   m  = 8 * half + r;
      ctx[((size_t)b * Sn + s0 + m) * Dn + h * DKn + nt * 16 + l16] =
          (__bf16)vv;
    }
  }
}

// ---------------------------------------------------------------------------
// Kernel 3: output projection  out = ctx @ Wout^T + bout   (f32 output)
// ---------------------------------------------------------------------------
__global__ __launch_bounds__(256) void mha_outproj(
    const __bf16* __restrict__ ctx, const __bf16* __restrict__ Woutb,
    const float* __restrict__ bout, float* __restrict__ out)
{
  int w    = blockIdx.x * 8 + (threadIdx.x >> 5);
  int lane = threadIdx.x & 31;
  int half = lane >> 4;
  int l16  = lane & 15;

  int n64   = w % (Dn / 64);  w /= (Dn / 64);
  int stile = w % (Sn / 16);
  int b     = w / (Sn / 16);
  int s0    = stile * 16;
  int n0    = n64 * 64;

  v8f zero = {};
  v8f acc[4];
  #pragma unroll
  for (int nt = 0; nt < 4; ++nt) acc[nt] = zero;

  const __bf16* arow = ctx + ((size_t)b * Sn + s0 + l16) * Dn;

  for (int k0 = 0; k0 < Dn; k0 += 32) {
    v16bf a;
    #pragma unroll
    for (int j = 0; j < 8; ++j) {
      a[j]     = arow[k0 + half * 16 + j];
      a[8 + j] = arow[k0 + half * 16 + 8 + j];
    }
    #pragma unroll
    for (int nt = 0; nt < 4; ++nt) {
      const v16bf* wp = (const v16bf*)(Woutb + (size_t)(n0 + nt * 16 + l16) * Dn
                                             + k0 + half * 16);
      acc[nt] = wmma_bf16(a, *wp, acc[nt]);
    }
  }

  #pragma unroll
  for (int nt = 0; nt < 4; ++nt) {
    float bv = bout[n0 + nt * 16 + l16];
    #pragma unroll
    for (int r = 0; r < 8; ++r) {
      int m = 8 * half + r;
      out[((size_t)b * Sn + s0 + m) * Dn + n0 + nt * 16 + l16] =
          acc[nt][r] + bv;
    }
  }
}

// ---------------------------------------------------------------------------
extern "C" void kernel_launch(void* const* d_in, const int* in_sizes, int n_in,
                              void* d_out, int out_size, void* d_ws,
                              size_t ws_size, hipStream_t stream) {
  (void)in_sizes; (void)n_in; (void)out_size; (void)ws_size;

  const float* q    = (const float*)d_in[0];
  const float* k    = (const float*)d_in[1];
  const float* v    = (const float*)d_in[2];
  const float* Wq   = (const float*)d_in[3];
  const float* bq   = (const float*)d_in[4];
  const float* Wout = (const float*)d_in[5];
  const float* bout = (const float*)d_in[6];
  float*       out  = (float*)d_out;

  // workspace layout (bf16 elements):
  //   xb   : 3*B*S*D      (q,k,v converted)         -- later reused as ctx
  //   qh   : B*H*S*DK
  //   kh   : B*H*S*DK
  //   vt   : B*H*S*DK     (transposed [b,h,e,s])
  //   Wqb  : H*DK*D
  //   Woutb: D*D
  // total ~49 MB
  const size_t NX = (size_t)Bn * Sn * Dn;
  const size_t NE = (size_t)Bn * Hn * Sn * DKn;
  const size_t NW = (size_t)Hn * DKn * Dn;
  __bf16* xb    = (__bf16*)d_ws;
  __bf16* qh    = xb + 3 * NX;
  __bf16* kh    = qh + NE;
  __bf16* vt    = kh + NE;
  __bf16* Wqb   = vt + NE;
  __bf16* Woutb = Wqb + NW;
  __bf16* ctx   = xb;   // alias: xb is dead after the projection stage

  dim3 blk(256);

  size_t cvt4 = (3 * NX + NW + (size_t)Dn * Dn) / 4;          // 3,276,800
  mha_convert<<<(int)(cvt4 / 256), blk, 0, stream>>>(q, k, v, Wq, Wout,
                                                     xb, Wqb, Woutb);

  int projWaves = 3 * Bn * Hn * (Sn / 16);                    // 12288
  mha_proj<<<projWaves / 8, blk, 0, stream>>>(xb, Wqb, bq, qh, kh, vt);

  int attnBlocks = Bn * Hn * (Sn / 128);                      // 512
  mha_attn<<<attnBlocks, blk, 0, stream>>>(qh, kh, vt, ctx);

  int outWaves = Bn * (Sn / 16) * (Dn / 64);                  // 4096
  mha_outproj<<<outWaves / 8, blk, 0, stream>>>(ctx, Woutb, bout, out);
}
